// LSTM_71734543778418
// MI455X (gfx1250) — matrix operand order repaired and checked by reading, not compiled, not run on hardware
//
#include <hip/hip_runtime.h>
#include <cstdint>

#define B_SZ 4096
#define D_SZ 2048
#define H_SZ 2048

#define BM 128
#define BN 128
#define BK 32
#define LDSS 40   // padded LDS row stride in bf16 elements (32 data + 8 pad = 80B, 16B aligned)

typedef __attribute__((ext_vector_type(16))) __bf16 bf16x16;
typedef __attribute__((ext_vector_type(8)))  float  floatx8;
typedef int v4i __attribute__((vector_size(16)));   // exact param type per hipcc diagnostic

#if defined(__has_builtin)
#  if __has_builtin(__builtin_amdgcn_global_load_async_to_lds_b128)
#    define HAVE_ASYNC_LDS 1
#  endif
#endif
#ifndef HAVE_ASYNC_LDS
#  define HAVE_ASYNC_LDS 0
#endif

#if HAVE_ASYNC_LDS
#  if defined(__has_builtin) && __has_builtin(__builtin_amdgcn_s_wait_asynccnt)
#    define WAIT_ASYNC(n) __builtin_amdgcn_s_wait_asynccnt(n)
#  else
#    define WAIT_ASYNC(n) asm volatile("s_wait_asynccnt %0" ::"i"(n) : "memory")
#  endif
#else
#  define WAIT_ASYNC(n) ((void)0)
#endif

union Frag {
  uint4   q[2];
  bf16x16 v;
};

__device__ __forceinline__ unsigned short f2bf(float f) {
  unsigned u = __builtin_bit_cast(unsigned, f);
  unsigned r = u + 0x7FFFu + ((u >> 16) & 1u);   // round-to-nearest-even
  return (unsigned short)(r >> 16);
}

// ---------------------------------------------------------------- fp32 -> bf16
__global__ __launch_bounds__(256) void cvt4_kernel(const float4* __restrict__ in,
                                                   ushort4* __restrict__ out, int n4) {
  int i = blockIdx.x * 256 + threadIdx.x;
  if (i < n4) {
    float4 f = in[i];
    out[i] = make_ushort4(f2bf(f.x), f2bf(f.y), f2bf(f.z), f2bf(f.w));
  }
}

// Stage one 128x32 bf16 tile of A and of W into LDS.
// 512 16B chunks per matrix; thread t handles chunks t and t+256.
__device__ __forceinline__ void stage_tile(
    const unsigned short* __restrict__ A, const unsigned short* __restrict__ W,
    unsigned short* lAbuf, unsigned short* lBbuf,
    int bm, int bn, int K, int k0, int tid)
{
  const int r0 = (tid)       >> 2;  const int c0 = ((tid)       & 3) * 8;
  const int r1 = (tid + 256) >> 2;  const int c1 = ((tid + 256) & 3) * 8;
#if HAVE_ASYNC_LDS
  __builtin_amdgcn_global_load_async_to_lds_b128(
      (v4i*)(A + (size_t)(bm + r0) * K + k0 + c0),
      (v4i*)(lAbuf + r0 * LDSS + c0), 0, 0);
  __builtin_amdgcn_global_load_async_to_lds_b128(
      (v4i*)(A + (size_t)(bm + r1) * K + k0 + c1),
      (v4i*)(lAbuf + r1 * LDSS + c1), 0, 0);
  __builtin_amdgcn_global_load_async_to_lds_b128(
      (v4i*)(W + (size_t)(bn + r0) * K + k0 + c0),
      (v4i*)(lBbuf + r0 * LDSS + c0), 0, 0);
  __builtin_amdgcn_global_load_async_to_lds_b128(
      (v4i*)(W + (size_t)(bn + r1) * K + k0 + c1),
      (v4i*)(lBbuf + r1 * LDSS + c1), 0, 0);
#else
  *(uint4*)&lAbuf[r0 * LDSS + c0] = *(const uint4*)(A + (size_t)(bm + r0) * K + k0 + c0);
  *(uint4*)&lAbuf[r1 * LDSS + c1] = *(const uint4*)(A + (size_t)(bm + r1) * K + k0 + c1);
  *(uint4*)&lBbuf[r0 * LDSS + c0] = *(const uint4*)(W + (size_t)(bn + r0) * K + k0 + c0);
  *(uint4*)&lBbuf[r1 * LDSS + c1] = *(const uint4*)(W + (size_t)(bn + r1) * K + k0 + c1);
#endif
}

// ---------------------------------------------------------------- fused GEMM
// Cout[M,N] = (Cin?) + sum_p Ap[M,K] @ Wp[N,K]^T + sum_p bp[N]
// bf16 operands, f32 accumulation via v_wmma_f32_16x16x32_bf16.
// Double-buffered LDS; async DMA staging overlapped with WMMA compute.
__global__ __launch_bounds__(256) void gemm_bf16_wmma(
    const unsigned short* __restrict__ A0, const unsigned short* __restrict__ W0, const float* __restrict__ b0,
    const unsigned short* __restrict__ A1, const unsigned short* __restrict__ W1, const float* __restrict__ b1,
    const unsigned short* __restrict__ A2, const unsigned short* __restrict__ W2, const float* __restrict__ b2,
    const float* __restrict__ Cin, float* __restrict__ Cout,
    int M, int N, int K)
{
  __shared__ unsigned short lA[2][BM * LDSS];
  __shared__ unsigned short lB[2][BN * LDSS];

  const int tid  = threadIdx.x;
  const int lane = tid & 31;
  const int wave = tid >> 5;
  const int wm   = wave >> 2;   // 0..1 : wave row (64 M each)
  const int wn   = wave & 3;    // 0..3 : wave col (32 N each)
  const int half = lane >> 4;   // 0/1  : lane half (K-split per ISA layout)
  const int lr   = lane & 15;

  const int bm = blockIdx.y * BM;
  const int bn = blockIdx.x * BN;

  floatx8 acc[4][2] = {};

  const unsigned short* As[3] = {A0, A1, A2};
  const unsigned short* Ws[3] = {W0, W1, W2};

  for (int p = 0; p < 3; ++p) {
    const unsigned short* A = As[p];
    const unsigned short* W = Ws[p];
    if (A == nullptr) continue;
    const int T = K / BK;

    stage_tile(A, W, lA[0], lB[0], bm, bn, K, 0, tid);

    for (int t = 0; t < T; ++t) {
      const int cur = t & 1;
      if (t + 1 < T) {
        // kick off next tile's DMA into the other buffer, then retire current tile
        stage_tile(A, W, lA[1 - cur], lB[1 - cur], bm, bn, K, (t + 1) * BK, tid);
        WAIT_ASYNC(4);
      } else {
        WAIT_ASYNC(0);
      }
      __syncthreads();

      const unsigned short* baseA = lA[cur];
      const unsigned short* baseB = lB[cur];

      // A fragment, 16-bit 16x32 layout: lanes 0-15 hold K{0..7,16..23}, lanes 16-31 K{8..15,24..31}
      Frag fa[4];
      #pragma unroll
      for (int i = 0; i < 4; ++i) {
        const unsigned short* pa = &baseA[(wm * 64 + i * 16 + lr) * LDSS + half * 8];
        fa[i].q[0] = *(const uint4*)(pa);
        fa[i].q[1] = *(const uint4*)(pa + 16);
      }
      // B fragment, 32x16: lanes 0-15 hold K{0..7,16..23} of column N=lane, lanes 16-31 K{8..15,24..31}
      Frag fb[2];
      #pragma unroll
      for (int j = 0; j < 2; ++j) {
        const unsigned short* pb = &baseB[(wn * 32 + j * 16 + lr) * LDSS + half * 8];
        fb[j].q[0] = *(const uint4*)(pb);
        fb[j].q[1] = *(const uint4*)(pb + 16);
      }

      #pragma unroll
      for (int i = 0; i < 4; ++i)
        #pragma unroll
        for (int j = 0; j < 2; ++j)
          acc[i][j] = __builtin_amdgcn_wmma_f32_16x16x32_bf16(
              false, fa[i].v, false, fb[j].v, (short)0, acc[i][j], false, false);

      __syncthreads();   // protect buffer `cur` from being restaged next iteration
    }
  }

  // epilogue: C/D layout: vgpr r -> M = r + 8*half, N = lane%16
  #pragma unroll
  for (int j = 0; j < 2; ++j) {
    const int col = bn + wn * 32 + j * 16 + lr;
    float bias = 0.0f;
    if (b0) bias += b0[col];
    if (b1) bias += b1[col];
    if (b2) bias += b2[col];
    #pragma unroll
    for (int i = 0; i < 4; ++i) {
      const int rowbase = bm + wm * 64 + i * 16 + half * 8;
      #pragma unroll
      for (int r = 0; r < 8; ++r) {
        const size_t idx = (size_t)(rowbase + r) * N + col;
        float v = acc[i][j][r] + bias;
        if (Cin) v += Cin[idx];
        Cout[idx] = v;
      }
    }
  }
}

// ---------------------------------------------------------------- elementwise
__global__ __launch_bounds__(256) void cellstate_kernel(
    const float* __restrict__ pre_i, const float* __restrict__ pre_f,
    const float* __restrict__ pre_c, const float* __restrict__ c_in,
    float* __restrict__ c_next, unsigned short* __restrict__ c_next_bf, size_t n)
{
  size_t i = (size_t)blockIdx.x * 256 + threadIdx.x;
  if (i >= n) return;
  float gi = 1.0f / (1.0f + __expf(-pre_i[i]));
  float gf = 1.0f / (1.0f + __expf(-pre_f[i]));
  float cn = gf * c_in[i] + gi * tanhf(pre_c[i]);
  c_next[i]    = cn;
  c_next_bf[i] = f2bf(cn);
}

__global__ __launch_bounds__(256) void output_kernel(
    const float* __restrict__ pre_o, const float* __restrict__ c_next,
    float* __restrict__ o_out, float* __restrict__ h_out, size_t n)
{
  size_t i = (size_t)blockIdx.x * 256 + threadIdx.x;
  if (i >= n) return;
  float t = tanhf(c_next[i]);
  float o = (1.0f / (1.0f + __expf(-pre_o[i]))) * t;  // reference: o already includes tanh(c')
  o_out[i] = o;
  h_out[i] = o * t;                                   // reference multiplies by tanh(c') again
}

// ---------------------------------------------------------------- launcher
extern "C" void kernel_launch(void* const* d_in, const int* in_sizes, int n_in,
                              void* d_out, int out_size, void* d_ws, size_t ws_size,
                              hipStream_t stream)
{
  (void)in_sizes; (void)n_in; (void)out_size; (void)ws_size;
  const int B = B_SZ, D = D_SZ, H = H_SZ;
  const size_t BD = (size_t)B * D, BH = (size_t)B * H;
  const size_t HD = (size_t)H * D, HH = (size_t)H * H;

  const float* x    = (const float*)d_in[0];
  const float* h    = (const float*)d_in[1];
  const float* c    = (const float*)d_in[2];
  const float* W_ix = (const float*)d_in[3];  const float* b_ix = (const float*)d_in[4];
  const float* W_fx = (const float*)d_in[5];  const float* b_fx = (const float*)d_in[6];
  const float* W_cx = (const float*)d_in[7];  const float* b_cx = (const float*)d_in[8];
  const float* W_ox = (const float*)d_in[9];  const float* b_ox = (const float*)d_in[10];
  const float* W_ih = (const float*)d_in[11]; const float* b_ih = (const float*)d_in[12];
  const float* W_ic = (const float*)d_in[13]; const float* b_ic = (const float*)d_in[14];
  const float* W_fh = (const float*)d_in[15]; const float* b_fh = (const float*)d_in[16];
  const float* W_fc = (const float*)d_in[17]; const float* b_fc = (const float*)d_in[18];
  const float* W_ch = (const float*)d_in[19]; const float* b_ch = (const float*)d_in[20];
  const float* W_oh = (const float*)d_in[21]; const float* b_oh = (const float*)d_in[22];
  const float* W_oc = (const float*)d_in[23]; const float* b_oc = (const float*)d_in[24];

  char* wsp = (char*)d_ws;
  size_t off = 0;
  auto carve = [&](size_t bytes) -> char* {
    char* p = wsp + off;
    off += (bytes + 255) & ~(size_t)255;
    return p;
  };

  unsigned short* xb   = (unsigned short*)carve(BD * 2);
  unsigned short* hb   = (unsigned short*)carve(BH * 2);
  unsigned short* cb   = (unsigned short*)carve(BH * 2);
  unsigned short* cnb  = (unsigned short*)carve(BH * 2);
  unsigned short* Wixb = (unsigned short*)carve(HD * 2);
  unsigned short* Wfxb = (unsigned short*)carve(HD * 2);
  unsigned short* Wcxb = (unsigned short*)carve(HD * 2);
  unsigned short* Woxb = (unsigned short*)carve(HD * 2);
  unsigned short* Wihb = (unsigned short*)carve(HH * 2);
  unsigned short* Wicb = (unsigned short*)carve(HH * 2);
  unsigned short* Wfhb = (unsigned short*)carve(HH * 2);
  unsigned short* Wfcb = (unsigned short*)carve(HH * 2);
  unsigned short* Wchb = (unsigned short*)carve(HH * 2);
  unsigned short* Wohb = (unsigned short*)carve(HH * 2);
  unsigned short* Wocb = (unsigned short*)carve(HH * 2);
  float* pre_i = (float*)carve(BH * 4);
  float* pre_f = (float*)carve(BH * 4);
  float* pre_c = (float*)carve(BH * 4);
  float* pre_o = (float*)carve(BH * 4);

  auto cvt = [&](const float* src, unsigned short* dst, size_t n) {
    int n4 = (int)(n / 4);
    cvt4_kernel<<<dim3((n4 + 255) / 256), dim3(256), 0, stream>>>(
        (const float4*)src, (ushort4*)dst, n4);
  };

  cvt(x, xb, BD);   cvt(h, hb, BH);   cvt(c, cb, BH);
  cvt(W_ix, Wixb, HD); cvt(W_fx, Wfxb, HD); cvt(W_cx, Wcxb, HD); cvt(W_ox, Woxb, HD);
  cvt(W_ih, Wihb, HH); cvt(W_ic, Wicb, HH); cvt(W_fh, Wfhb, HH); cvt(W_fc, Wfcb, HH);
  cvt(W_ch, Wchb, HH); cvt(W_oh, Wohb, HH); cvt(W_oc, Wocb, HH);

  dim3 gg(H / BN, B / BM);   // (16, 32)
  dim3 bb(256);

  // gate i: x@W_ix^T + h@W_ih^T + c@W_ic^T + biases
  gemm_bf16_wmma<<<gg, bb, 0, stream>>>(xb, Wixb, b_ix, hb, Wihb, b_ih, cb, Wicb, b_ic,
                                        nullptr, pre_i, B, H, D);
  // gate f
  gemm_bf16_wmma<<<gg, bb, 0, stream>>>(xb, Wfxb, b_fx, hb, Wfhb, b_fh, cb, Wfcb, b_fc,
                                        nullptr, pre_f, B, H, D);
  // candidate c̄
  gemm_bf16_wmma<<<gg, bb, 0, stream>>>(xb, Wcxb, b_cx, hb, Wchb, b_ch,
                                        nullptr, nullptr, nullptr,
                                        nullptr, pre_c, B, H, D);
  // o partial (x,h terms)
  gemm_bf16_wmma<<<gg, bb, 0, stream>>>(xb, Woxb, b_ox, hb, Wohb, b_oh,
                                        nullptr, nullptr, nullptr,
                                        nullptr, pre_o, B, H, D);

  float* out   = (float*)d_out;
  float* o_out = out;
  float* h_out = out + BH;
  float* c_nxt = out + 2 * BH;

  cellstate_kernel<<<dim3((unsigned)(BH / 256)), bb, 0, stream>>>(
      pre_i, pre_f, pre_c, c, c_nxt, cnb, BH);

  // o += c_next @ W_oc^T + b_oc   (accumulate onto pre_o via Cin)
  gemm_bf16_wmma<<<gg, bb, 0, stream>>>(cnb, Wocb, b_oc,
                                        nullptr, nullptr, nullptr,
                                        nullptr, nullptr, nullptr,
                                        pre_o, pre_o, B, H, H);

  output_kernel<<<dim3((unsigned)(BH / 256)), bb, 0, stream>>>(
      pre_o, c_nxt, o_out, h_out, BH);
}
